// Retention_10737418240797
// MI455X (gfx1250) — matrix operand order repaired and checked by reading, not compiled, not run on hardware
//
#include <hip/hip_runtime.h>

// ---------------------------------------------------------------------------
// Retention forward for MI455X (gfx1250), wave32 + WMMA bf16.
//   out[b,q,:] = sum_k gamma^(q-k) * (Q[q]·K[k]) * V[k]   (causal)
// Pipeline:
//   K0 wconv    : W_Q|W_K|W_V (f32) -> WcT bf16 transposed [1024][2048]
//   K1 proj_xpos: QKV projection GEMM (bf16 WMMA) + xpos; sched_group_barrier
//                 enforced load/WMMA software pipeline
//   K2 retention: causal decayed attention, bf16 WMMA, per-wave LDS transpose
//                 of P, sched_group_barrier enforced pipeline
// ---------------------------------------------------------------------------

typedef __attribute__((ext_vector_type(16))) __bf16 v16bf;
typedef __attribute__((ext_vector_type(8)))  __bf16 v8bf;
typedef __attribute__((ext_vector_type(8)))  float  v8f;
typedef __attribute__((ext_vector_type(4)))  float  v4f;

union BF16x16 { v16bf v; v8bf h[2]; unsigned short u[16]; };

static constexpr int HIDDEN = 2048;
static constexpr int HEAD   = 256;
static constexpr int VDIM   = 512;
static constexpr int SEQ    = 4096;
static constexpr int BATCH  = 4;
static constexpr int NCOLS  = HEAD + HEAD + VDIM;   // 1024

// sched_group_barrier masks: 0x8 = MFMA/WMMA, 0x20 = VMEM read, 0x100 = DS read
#if __has_builtin(__builtin_amdgcn_sched_group_barrier)
#define SGB(mask, size) __builtin_amdgcn_sched_group_barrier((mask), (size), 0)
#else
#define SGB(mask, size)
#endif

__device__ __forceinline__ unsigned short f2bf_u16(float f) {
    unsigned u = __builtin_bit_cast(unsigned, f);
    // round-to-nearest-even truncation to bf16
    return (unsigned short)((u + 0x7FFFu + ((u >> 16) & 1u)) >> 16);
}

__device__ __forceinline__ v8f wmma_bf16(v16bf a, v16bf b, v8f c) {
    // emits v_wmma_f32_16x16x32_bf16
    return __builtin_amdgcn_wmma_f32_16x16x32_bf16(
        /*neg_a=*/false, a, /*neg_b=*/false, b,
        /*c_mod=*/(short)0, c, /*reuse_a=*/false, /*reuse_b=*/false);
}

// ---------------------------------------------------------------------------
// K0: build transposed bf16 weight matrix WcT[j][h], j in [0,1024)
// ---------------------------------------------------------------------------
__global__ __launch_bounds__(256) void wconv_kernel(
    const float* __restrict__ WQ, const float* __restrict__ WK,
    const float* __restrict__ WV, unsigned short* __restrict__ WcT)
{
    int idx = blockIdx.x * 256 + threadIdx.x;     // 0 .. 1024*2048-1
    int j = idx >> 11;                            // col (output feature)
    int h = idx & (HIDDEN - 1);                   // hidden index
    float v;
    if (j < 256)       v = WQ[(size_t)h * HEAD + j];
    else if (j < 512)  v = WK[(size_t)h * HEAD + (j - 256)];
    else               v = WV[(size_t)h * VDIM + (j - 512)];
    WcT[idx] = f2bf_u16(v);
}

// ---------------------------------------------------------------------------
// K1: projection GEMM + xpos.
// Grid: 512 blocks of 256 threads. block = (mb 0..127, cg 0..3).
// Wave w owns rows [mb*128 + w*16, +16), cols [cg*256, +256).
// ---------------------------------------------------------------------------
__global__ __launch_bounds__(256) void proj_xpos_kernel(
    const float* __restrict__ X, const unsigned short* __restrict__ WcT,
    unsigned short* __restrict__ Qs, unsigned short* __restrict__ Ks,
    unsigned short* __restrict__ Vt)
{
    const int lane = threadIdx.x & 31;
    const int wave = threadIdx.x >> 5;
    const int lhi  = lane >> 4;       // 0/1: lane half (k-offset group)
    const int llo  = lane & 15;       // row (A) / col (B,C,D) within tile
    const int mb   = blockIdx.x >> 2;
    const int cg   = blockIdx.x & 3;
    const int m0   = mb * 128 + wave * 16;
    const int colbase = cg * 256;

    const v8f vzero = {0.f,0.f,0.f,0.f,0.f,0.f,0.f,0.f};
    v8f acc[16];
    #pragma unroll
    for (int t = 0; t < 16; ++t) acc[t] = vzero;

    const float* xrow = X + (size_t)(m0 + llo) * HIDDEN;

    auto loadB = [&](BF16x16& bm, int t, int kc) {
        const unsigned short* wp =
            WcT + (size_t)(colbase + t * 16 + llo) * HIDDEN + kc * 32 + 8 * lhi;
        bm.h[0] = *(const v8bf*)(wp);
        bm.h[1] = *(const v8bf*)(wp + 16);
    };

    // A-operand (X) floats, prefetched one k-chunk ahead
    v4f f0, f1, f2, f3;
    {
        const float* xp = xrow + 8 * lhi;
        f0 = *(const v4f*)(xp);      f1 = *(const v4f*)(xp + 4);
        f2 = *(const v4f*)(xp + 16); f3 = *(const v4f*)(xp + 20);
    }

    for (int kc = 0; kc < HIDDEN / 32; ++kc) {
        // convert current A floats -> bf16 A tile (VALU, co-executes with WMMA)
        BF16x16 a;
        #pragma unroll
        for (int e = 0; e < 4; ++e) {
            a.u[e]      = f2bf_u16(f0[e]);
            a.u[4 + e]  = f2bf_u16(f1[e]);
            a.u[8 + e]  = f2bf_u16(f2[e]);
            a.u[12 + e] = f2bf_u16(f3[e]);
        }
        // issue next chunk's A loads (branchless wrap keeps body one region)
        v4f g0, g1, g2, g3;
        {
            const int kcn = (kc + 1) & (HIDDEN / 32 - 1);
            const float* xp = xrow + kcn * 32 + 8 * lhi;
            g0 = *(const v4f*)(xp);      g1 = *(const v4f*)(xp + 4);
            g2 = *(const v4f*)(xp + 16); g3 = *(const v4f*)(xp + 20);
        }

        // B tiles, 4-deep rotating buffer; schedule pinned below
        BF16x16 bm[4];
        loadB(bm[0], 0, kc);
        loadB(bm[1], 1, kc);
        loadB(bm[2], 2, kc);
        #pragma unroll
        for (int t = 0; t < 16; ++t) {
            if (t + 3 < 16) loadB(bm[(t + 3) & 3], t + 3, kc);
            acc[t] = wmma_bf16(a.v, bm[t & 3].v, acc[t]);
        }

        f0 = g0; f1 = g1; f2 = g2; f3 = g3;

        // pinned pipeline: 10 loads ahead (4 A-next + 3 B tiles),
        // then [1 WMMA, 2 loads] x13, then 3 WMMA tail
        SGB(0x20, 10);
        #pragma unroll
        for (int i = 0; i < 13; ++i) { SGB(0x8, 1); SGB(0x20, 2); }
        SGB(0x8, 3);
    }

    // ---- epilogue ----
    const int b     = m0 / SEQ;
    const int sbase = m0 & (SEQ - 1);

    if (cg <= 1) {
        unsigned short* outp = (cg == 0) ? Qs : Ks;
        const float signp = (cg == 0) ? 1.f : -1.f;     // K uses 1/scale
        #pragma unroll
        for (int t = 0; t < 16; ++t) {
            const int j = t * 16 + llo;                 // head dim 0..255
            const int i = j >> 1;                       // pair index 0..127
            // per-frequency constants hoisted out of the row loop
            float sv   = (2.f * (float)i + 102.4f) * (1.f / 358.4f);
            float lsv  = __builtin_amdgcn_logf(sv) * (signp / 512.f);
            float invf = __builtin_amdgcn_exp2f(-(float)i * (13.287712379549449f / 128.f))
                         * 0.15915494309189535f;        // revolutions per position
            #pragma unroll
            for (int r = 0; r < 8; ++r) {
                const int s = sbase + r + 8 * lhi;      // seq position
                float scl = __builtin_amdgcn_exp2f((float)s * lsv);
                float rev = (float)s * invf;
                rev -= __builtin_floorf(rev);
                float sn = __builtin_amdgcn_sinf(rev);
                float cs = __builtin_amdgcn_cosf(rev);
                float aV = acc[t][r];
                float pV = __shfl_xor(aV, 1, 32);       // partner column j^1
                float rot = (j & 1) ? pV : -pV;         // rotate_every_two
                float out = (aV * cs + rot * sn) * scl;
                outp[(size_t)(m0 + r + 8 * lhi) * HEAD + j] = f2bf_u16(out);
            }
        }
    } else {
        const int vc0 = (cg - 2) * 256;
        #pragma unroll
        for (int t = 0; t < 16; ++t) {
            #pragma unroll
            for (int r = 0; r < 8; ++r) {
                const int vcol = vc0 + t * 16 + llo;
                const int s    = sbase + r + 8 * lhi;
                Vt[((size_t)(b * VDIM + vcol)) * SEQ + s] = f2bf_u16(acc[t][r]);
            }
        }
    }
}

// ---------------------------------------------------------------------------
// K2: causal decayed attention. 2048 independent waves (256 blocks x 8).
// Wave owns (batch b, 16 queries q0..q0+15, 256 V-cols vbase..+255).
// ---------------------------------------------------------------------------
__global__ __launch_bounds__(256) void retention_kernel(
    const unsigned short* __restrict__ Qs, const unsigned short* __restrict__ Ks,
    const unsigned short* __restrict__ Vt, float* __restrict__ Out)
{
    __shared__ unsigned short ldsP[8 * 16 * 32];   // per-wave 16x32 P tile (1 KB)

    const int lane = threadIdx.x & 31;
    const int wave = threadIdx.x >> 5;
    const int lhi  = lane >> 4;
    const int llo  = lane & 15;

    const int wg    = blockIdx.x * 8 + wave;       // 0..2047
    const int b     = wg >> 9;                     // 512 waves per batch
    const int qt    = (wg & 511) >> 1;
    const int vh    = wg & 1;
    const int q0    = qt * 16;
    const int vbase = vh * 256;
    unsigned short* myP = ldsP + wave * 512;

    // Q A-tiles (16 queries x 256 head dims) held in registers
    v16bf qa[8];
    {
        const unsigned short* qrow = Qs + (size_t)(b * SEQ + q0 + llo) * HEAD;
        #pragma unroll
        for (int c = 0; c < 8; ++c) {
            BF16x16 t;
            t.h[0] = *(const v8bf*)(qrow + c * 32 + 8 * lhi);
            t.h[1] = *(const v8bf*)(qrow + c * 32 + 16 + 8 * lhi);
            qa[c] = t.v;
        }
    }

    const v8f vzero = {0.f,0.f,0.f,0.f,0.f,0.f,0.f,0.f};
    v8f acc[16];
    #pragma unroll
    for (int t = 0; t < 16; ++t) acc[t] = vzero;

    const float LOG2G = -0.04580368961312478f;     // log2(31/32)
    const int kb_end   = (q0 + 15) >> 5;           // inclusive, 32-key blocks
    const int kb_start = (q0 > 2816) ? ((q0 - 2816) >> 5) : 0; // gamma^d underflow window

    for (int kb = kb_start; kb <= kb_end; ++kb) {
        const int k0 = kb * 32;

        // ---- S = Q K^T over this 32-key block (two 16-key tiles) ----
        const unsigned short* krow0 =
            Ks + (size_t)(b * SEQ + k0 + llo) * HEAD + 8 * lhi;
        const unsigned short* krow1 = krow0 + (size_t)16 * HEAD;

        auto loadK = [&](BF16x16& bm, const unsigned short* base, int c) {
            bm.h[0] = *(const v8bf*)(base + c * 32);
            bm.h[1] = *(const v8bf*)(base + c * 32 + 16);
        };

        v8f s0 = vzero, s1 = vzero;
        BF16x16 kt0[2], kt1[2];
        loadK(kt0[0], krow0, 0);
        loadK(kt1[0], krow1, 0);
        #pragma unroll
        for (int c = 0; c < 8; ++c) {
            if (c + 1 < 8) {
                loadK(kt0[(c + 1) & 1], krow0, c + 1);
                loadK(kt1[(c + 1) & 1], krow1, c + 1);
            }
            s0 = wmma_bf16(qa[c], kt0[c & 1].v, s0);
            s1 = wmma_bf16(qa[c], kt1[c & 1].v, s1);
        }

        // V B-tile addressing; first V load issued EARLY (overlaps decay+LDS)
        const unsigned short* vcol0 =
            Vt + ((size_t)(b * VDIM + vbase + llo)) * SEQ + k0 + 8 * lhi;
        auto loadV = [&](BF16x16& bm, int t) {
            const unsigned short* vp = vcol0 + (size_t)t * 16 * SEQ;
            bm.h[0] = *(const v8bf*)(vp);
            bm.h[1] = *(const v8bf*)(vp + 16);
        };
        BF16x16 vb[2];
        loadV(vb[0], 0);

        // ---- decay * mask, f32 -> bf16, C-layout -> LDS [query][key] ----
        #pragma unroll
        for (int r = 0; r < 8; ++r) {
            const int q  = q0 + r + 8 * lhi;
            const int d0 = q - (k0 + llo);
            const int d1 = d0 - 16;
            float p0 = (d0 < 0) ? 0.f
                     : s0[r] * __builtin_amdgcn_exp2f((float)d0 * LOG2G);
            float p1 = (d1 < 0) ? 0.f
                     : s1[r] * __builtin_amdgcn_exp2f((float)d1 * LOG2G);
            myP[(r + 8 * lhi) * 32 + llo]      = f2bf_u16(p0);
            myP[(r + 8 * lhi) * 32 + 16 + llo] = f2bf_u16(p1);
        }
        asm volatile("s_wait_dscnt 0" ::: "memory");   // per-wave LDS RAW fence

        // reload P as A-tile (16 queries x 32 keys)
        BF16x16 pa;
        pa.h[0] = *(const v8bf*)(myP + llo * 32 + 8 * lhi);
        pa.h[1] = *(const v8bf*)(myP + llo * 32 + 16 + 8 * lhi);

        // ---- acc += P V over 16 column tiles, double-buffered ----
        #pragma unroll
        for (int t = 0; t < 16; ++t) {
            if (t + 1 < 16) loadV(vb[(t + 1) & 1], t + 1);
            acc[t] = wmma_bf16(pa.v, vb[t & 1].v, acc[t]);
        }

        // pinned pipeline for the whole block body:
        //  S-phase: 8 K-loads ahead, [2 WMMA, 4 loads] x6, 4 WMMA tail
        SGB(0x20, 8);
        #pragma unroll
        for (int i = 0; i < 6; ++i) { SGB(0x8, 2); SGB(0x20, 4); }
        SGB(0x8, 4);
        //  first V tile overlaps decay VALU + DS stores
        SGB(0x20, 2);
        //  PV-phase: P reload (DS), V t=1, then [1 WMMA, 2 loads] x14, 2 WMMA
        SGB(0x100, 2);
        SGB(0x20, 2);
        #pragma unroll
        for (int i = 0; i < 14; ++i) { SGB(0x8, 1); SGB(0x20, 2); }
        SGB(0x8, 2);
    }

    // store f32 output [b][q][vdim]
    #pragma unroll
    for (int t = 0; t < 16; ++t) {
        #pragma unroll
        for (int r = 0; r < 8; ++r) {
            Out[(size_t)(b * SEQ + q0 + r + 8 * lhi) * VDIM + vbase + t * 16 + llo]
                = acc[t][r];
        }
    }
}

// ---------------------------------------------------------------------------
extern "C" void kernel_launch(void* const* d_in, const int* in_sizes, int n_in,
                              void* d_out, int out_size, void* d_ws, size_t ws_size,
                              hipStream_t stream)
{
    const float* X  = (const float*)d_in[0];
    const float* WQ = (const float*)d_in[1];
    const float* WK = (const float*)d_in[2];
    const float* WV = (const float*)d_in[3];
    float* Out = (float*)d_out;

    // workspace layout (bf16 bit-pattern arrays), total 36 MB
    unsigned short* WcT = (unsigned short*)d_ws;               // 1024*2048 = 2M
    unsigned short* Qsb = WcT + (size_t)NCOLS * HIDDEN;        // 16384*256 = 4M
    unsigned short* Ksb = Qsb + (size_t)BATCH * SEQ * HEAD;    // 4M
    unsigned short* Vtb = Ksb + (size_t)BATCH * SEQ * HEAD;    // 4*512*4096 = 8M

    // K0: weights -> transposed bf16
    wconv_kernel<<<(NCOLS * HIDDEN) / 256, 256, 0, stream>>>(WQ, WK, WV, WcT);

    // K1: projections + xpos. 128 M-blocks x 4 col-groups.
    proj_xpos_kernel<<<512, 256, 0, stream>>>(X, WcT, Qsb, Ksb, Vtb);

    // K2: retention. 2048 independent waves, 8 per block.
    retention_kernel<<<256, 256, 0, stream>>>(Qsb, Ksb, Vtb, Out);

    (void)in_sizes; (void)n_in; (void)out_size; (void)ws_size;
}
